// MPLayer_24799141167507
// MI455X (gfx1250) — matrix operand order repaired and checked by reading, not compiled, not run on hardware
//
#include <hip/hip_runtime.h>

// MPLayer fused kernel for gfx1250 (MI455X), wave32 + WMMA f16 (f32 accum) +
// TDM (tensor_load_to_lds) staging of the edges stream.
//
// out[i,m] = inv_deg[i] * sum_{j,l,n} edges[i,j,n] * nodes[nlist[i,j],l] * w[l,m,n]
// Stage 1 (per node):  t[l,n]   = sum_j sliced[j,l]*edges[j,n]      (128x32 @ 32x8)
// Stage 2 (per tile):  out[i,m] = inv[i] * sum_k t[i,k]*W[k,m]      (16x1024 @ 1024x128)
// with k = l*8+n so W[k,m] = w[l,m,n] reads contiguously from w's (F,F,E) layout.

typedef _Float16     v16h  __attribute__((ext_vector_type(16)));
typedef _Float16     v8h   __attribute__((ext_vector_type(8)));
typedef __fp16       p2h   __attribute__((ext_vector_type(2)));  // cvt_pkrtz result
typedef float        v8f   __attribute__((ext_vector_type(8)));
typedef float        v4f   __attribute__((ext_vector_type(4)));
typedef unsigned int u32x4 __attribute__((ext_vector_type(4)));
typedef int          i32x8 __attribute__((ext_vector_type(8)));
typedef int          i32x4 __attribute__((ext_vector_type(4)));

#define N_NODES   50000
#define KNBR      32
#define FDIM      128
#define EDIM      8
#define NUM_TILES (N_NODES / 16)   // 3125 (exact)
#define WE        (FDIM * EDIM)    // 1024 = flattened contraction length
#define NT_STRIDE 40               // padded halves per feature row of gathered tile
#define T_STRIDE  1032             // padded halves per node row of t
#define ET_ELEMS  (16 * KNBR * EDIM)  // 4096 f32 edge elements per 16-node tile

#if __has_builtin(__builtin_amdgcn_tensor_load_to_lds)
#define HAVE_TDM 1
#else
#define HAVE_TDM 0
#endif

static __device__ __forceinline__ v16h cat8(v8h lo, v8h hi) {
  return __builtin_shufflevector(lo, hi, 0,1,2,3,4,5,6,7,8,9,10,11,12,13,14,15);
}

// 16 f32 -> v16h via v_cvt_pk_rtz_f16_f32 (8 packed converts)
static __device__ __forceinline__ v16h pack16(const float* f) {
  v16h r;
#pragma unroll
  for (int q = 0; q < 8; ++q) {
    p2h p = __builtin_amdgcn_cvt_pkrtz(f[2 * q], f[2 * q + 1]);
    r[2 * q]     = (_Float16)p[0];
    r[2 * q + 1] = (_Float16)p[1];
  }
  return r;
}

__global__ __launch_bounds__(256, 1)
void MPLayer_24799141167507_kernel(const float* __restrict__ nodes,
                                   const int*   __restrict__ nlist,
                                   const float* __restrict__ edges,
                                   const float* __restrict__ inv_degree,
                                   const float* __restrict__ w,
                                   float* __restrict__ out) {
  // Wave-private gathered-neighbor scratch (transposed: [feature][j]), 8 waves.
  __shared__ _Float16 sNT[8][FDIM * NT_STRIDE];   // 8 * 5120 halves = 80 KB
  __shared__ _Float16 sT[16][T_STRIDE];           // 16 * 1032 halves ~ 32 KB
  __shared__ float    sE[16][KNBR * EDIM];        // edges tile (f32), 16 KB
  __shared__ float    sInv[16];

  const int tid  = threadIdx.x;
  const int wv   = tid >> 5;        // wave id 0..7
  const int lane = tid & 31;
  const int col  = lane & 15;       // lane % 16 (WMMA column / M-row index)
  const int hi16 = lane >> 4;       // 0: lanes 0-15, 1: lanes 16-31
  const int offA = hi16 ? 8 : 0;    // A-operand K-offset pattern (f16 layout)

  for (int tile = blockIdx.x; tile < NUM_TILES; tile += gridDim.x) {
    if (tid < 16) sInv[tid] = inv_degree[tile * 16 + tid];

    // ---- Stage edges tile (16 nodes x 32 x 8 f32, contiguous 16 KB) into LDS.
#if HAVE_TDM
    if (wv == 0) {
      // Tensor DMA descriptor (D#), per CDNA5 ISA ch.8 bitfields.
      const unsigned long long gaddr =
          (unsigned long long)(const void*)(edges + (size_t)tile * ET_ELEMS);
      const unsigned int laddr = (unsigned int)(size_t)(void*)&sE[0][0];
      u32x4 g0;
      g0[0] = 1u;                                     // count=1, user descriptor
      g0[1] = laddr;                                  // lds_addr (bytes)
      g0[2] = (unsigned int)(gaddr & 0xFFFFFFFFull);  // global_addr[31:0]
      g0[3] = (unsigned int)((gaddr >> 32) & 0x1FFFFFFull) | (2u << 30);  // hi + type=2
      i32x8 g1;
      g1[0] = (int)(2u << 16);                 // wg_mask=0, data_size=4B
      g1[1] = (int)((ET_ELEMS & 0xFFFF) << 16);   // tensor_dim0[15:0] @ bits63:48
      g1[2] = (int)(((ET_ELEMS >> 16) & 0xFFFF) | (1u << 16)); // dim0 hi, tensor_dim1=1
      g1[3] = (int)((ET_ELEMS & 0xFFFF) << 16);   // tile_dim0 = 4096 @ bits127:112
      g1[4] = 1;                                   // tile_dim1 = 1
      g1[5] = ET_ELEMS;                            // tensor_dim0_stride lo
      g1[6] = 0;                                   // stride hi / dim1_stride lo
      g1[7] = 0;
      i32x4 z4 = {0, 0, 0, 0};
      i32x8 z8 = {0, 0, 0, 0, 0, 0, 0, 0};
      __builtin_amdgcn_tensor_load_to_lds(g0, g1, z4, z4, z8, 0);
    }
#else
    {
      const float* esrc = edges + (size_t)tile * ET_ELEMS;
      float*       edst = &sE[0][0];
#pragma unroll
      for (int q = 0; q < 4; ++q) {
        const int i4 = (tid + q * 256) * 4;
        *(v4f*)(edst + i4) = *(const v4f*)(esrc + i4);
      }
    }
#endif

    // ================= Stage 1: 2 nodes per wave =================
    for (int s = 0; s < 2; ++s) {
      const int nloc = 2 * wv + s;
      const int i    = tile * 16 + nloc;

      // Gather 32 neighbor feature rows, f32->f16, store TRANSPOSED into sNT[wv].
      // lane = neighbor j; iterate feature chunks of 4.
      const long long idx  = (long long)nlist[(size_t)i * KNBR + lane];
      const float*    nrow = nodes + (size_t)idx * FDIM;
      _Float16*       nt   = &sNT[wv][0];
#pragma unroll 8
      for (int c = 0; c < 32; ++c) {
        v4f v   = *(const v4f*)(nrow + 4 * c);
        p2h h01 = __builtin_amdgcn_cvt_pkrtz(v.x, v.y);
        p2h h23 = __builtin_amdgcn_cvt_pkrtz(v.z, v.w);
        nt[(4 * c + 0) * NT_STRIDE + lane] = (_Float16)h01[0];
        nt[(4 * c + 1) * NT_STRIDE + lane] = (_Float16)h01[1];
        nt[(4 * c + 2) * NT_STRIDE + lane] = (_Float16)h23[0];
        nt[(4 * c + 3) * NT_STRIDE + lane] = (_Float16)h23[1];
      }
#if HAVE_TDM
      if (s == 0 && wv == 0) __builtin_amdgcn_s_wait_tensorcnt(0);
#endif
      __syncthreads();

      // B operand: edges as K=32 x N=16 (columns n>=8 are zero padding),
      // sourced from the TDM-staged LDS tile.
      // B k order here (contiguous, same as A): lanes 0-15 hold K=0..15, lanes 16-31 hold K=16..31.
      float ef[16];
      const int j0 = hi16 ? 16 : 0;
#pragma unroll
      for (int jj = 0; jj < 16; ++jj) {
        float e = 0.0f;
        if (col < EDIM) e = sE[nloc][(j0 + jj) * EDIM + col];
        ef[jj] = e;
      }
      const v16h Be = pack16(ef);

      // A operand per 16-feature tile: sliced^T from transposed LDS image.
      // A k order here (contiguous, same as B): lanes 0-15 hold K=0..15, lanes 16-31 hold K=16..31.
#pragma unroll
      for (int lt = 0; lt < 8; ++lt) {
        const int f  = lt * 16 + col;
        v8h a0 = *(const v8h*)&nt[f * NT_STRIDE + (hi16 ? 16 : 0)];
        v8h a1 = *(const v8h*)&nt[f * NT_STRIDE + (hi16 ? 16 : 0) + 8];
        v16h Aop = cat8(a0, a1);
        v8f z = {};
        v8f d = __builtin_amdgcn_wmma_f32_16x16x32_f16(
            false, Aop, false, Be, (short)0, z, false, false);
        // D layout: lane col = n, VGPR r -> M = lt*16 + r (+8 for lanes 16-31).
        if (col < EDIM) {
#pragma unroll
          for (int r = 0; r < 8; ++r) {
            const int l = lt * 16 + r + (hi16 ? 8 : 0);
            sT[nloc][l * EDIM + col] = (_Float16)d[r];
          }
        }
      }
      __syncthreads();
    }

    // ================= Stage 2: one 16-feature output tile per wave =========
    // A = t (M=node, K=1024 in 32 steps), B = w slice (contiguous f32 runs).
    const int m = wv * 16 + col;   // output feature
    v8f acc = {};
#pragma unroll 4
    for (int ks = 0; ks < 32; ++ks) {
      v8h a0 = *(const v8h*)&sT[col][32 * ks + (hi16 ? 16 : 0)];
      v8h a1 = *(const v8h*)&sT[col][32 * ks + (hi16 ? 16 : 0) + 8];
      v16h Aop = cat8(a0, a1);

      // B: lane covers k = 32*ks + (hi16?16:0) + 0..15
      //    -> w rows l0 and l0+1, all n=0..7, feature m: two contiguous
      //       8-float runs, 32B apart across lanes (L2-resident, 512 KB).
      const int    l0 = 4 * ks + (hi16 ? 2 : 0);
      const float* wp = w + (size_t)l0 * WE + (size_t)m * EDIM;
      v4f b0 = *(const v4f*)(wp);
      v4f b1 = *(const v4f*)(wp + 4);
      v4f b2 = *(const v4f*)(wp + WE);
      v4f b3 = *(const v4f*)(wp + WE + 4);
      float bf[16] = {b0.x, b0.y, b0.z, b0.w, b1.x, b1.y, b1.z, b1.w,
                      b2.x, b2.y, b2.z, b2.w, b3.x, b3.y, b3.z, b3.w};
      v16h Bw = pack16(bf);

      acc = __builtin_amdgcn_wmma_f32_16x16x32_f16(
          false, Aop, false, Bw, (short)0, acc, false, false);
    }

    // Epilogue: D lane col = m_local, VGPR r -> node r (+8 for lanes 16-31).
#pragma unroll
    for (int r = 0; r < 8; ++r) {
      const int nloc = r + (hi16 ? 8 : 0);
      out[(size_t)(tile * 16 + nloc) * FDIM + m] = acc[r] * sInv[nloc];
    }
    __syncthreads();
  }
}

extern "C" void kernel_launch(void* const* d_in, const int* in_sizes, int n_in,
                              void* d_out, int out_size, void* d_ws, size_t ws_size,
                              hipStream_t stream) {
  const float* nodes = (const float*)d_in[0];
  const int*   nlist = (const int*)d_in[1];   // integer input per harness contract
  const float* edges = (const float*)d_in[2];
  const float* invd  = (const float*)d_in[3];
  const float* w     = (const float*)d_in[4];
  float*       out   = (float*)d_out;
  (void)in_sizes; (void)n_in; (void)out_size; (void)d_ws; (void)ws_size;

  dim3 grid(512), block(256);
  hipLaunchKernelGGL(MPLayer_24799141167507_kernel, grid, block, 0, stream,
                     nodes, nlist, edges, invd, w, out);
}
